// TGATClassifier_26637387169862
// MI455X (gfx1250) — compile-verified, matrix operand order checked by reading, not run on hardware
//
#include <hip/hip_runtime.h>
#include <hip/hip_bf16.h>
#include <math.h>

// ---- problem constants (match reference) ----
#define HEADS   2
#define HID     128
#define FIN     128
#define BQ      4
#define TT      8
#define NN      10000
#define EE      320000
#define OUTC    10
#define NEG_SLOPE 0.2f
#define C1      (HEADS*HID)          // 256
#define NPAD    10112                // multiple of 128 >= NN
#define ETOT    ((EE+NN)*HEADS)      // edges incl self-loops, per head

typedef __attribute__((ext_vector_type(16))) __bf16 bf16x16;
typedef __attribute__((ext_vector_type(8)))  __bf16 bf16x8;
typedef __attribute__((ext_vector_type(8)))  float  f32x8;

__device__ __forceinline__ __bf16 f2bf(float f) {
  unsigned u = __float_as_uint(f);
  u += 0x7FFFu + ((u >> 16) & 1u);     // round-to-nearest-even
  unsigned short h = (unsigned short)(u >> 16);
  __bf16 r;
  __builtin_memcpy(&r, &h, 2);
  return r;
}

__device__ __forceinline__ void atomicMaxF(float* addr, float val) {
  // ordered-float trick: positive -> signed int max; negative -> unsigned min
  if (val >= 0.f) atomicMax((int*)addr, __float_as_int(val));
  else            atomicMin((unsigned int*)addr, __float_as_uint(val));
}

// ---- convert one [N,FIN] f32 slice to bf16 with zero row padding ----
__global__ void k_convert_x(const float* __restrict__ x, __bf16* __restrict__ xb) {
  long i = (long)blockIdx.x * blockDim.x + threadIdx.x;
  long total = (long)NPAD * FIN;
  if (i >= total) return;
  long r = i / FIN, c = i % FIN;
  xb[i] = (r < NN) ? f2bf(x[r * FIN + c]) : f2bf(0.f);
}

// ---- transpose + bf16-convert weights: W[K,Ncol] -> Wt[Ncol,K] ----
__global__ void k_transpose_w(const float* __restrict__ W, __bf16* __restrict__ Wt,
                              int K, int Ncol) {
  int i = blockIdx.x * blockDim.x + threadIdx.x;
  if (i >= K * Ncol) return;
  int n = i / K, k = i % K;
  Wt[i] = f2bf(W[k * Ncol + n]);
}

// ---- bf16 WMMA GEMM: C[NPAD,256] = A[NPAD,K] @ Bt[256,K]^T, f32 accum ----
// block = 256 threads = 8 waves; wave computes 16 rows x 64 cols; block tile 128x64
__global__ void __launch_bounds__(256)
k_gemm_bf16(const __bf16* __restrict__ A, const __bf16* __restrict__ Bt,
            float* __restrict__ C, int K) {
  const int lane  = threadIdx.x & 31;
  const int wave  = threadIdx.x >> 5;
  const int row0  = blockIdx.x * 128 + wave * 16;
  const int col0  = blockIdx.y * 64;
  const int rlo   = lane & 15;
  const int kHalf = (lane >> 4) << 3;          // 0 for lanes 0-15, 8 for 16-31

  f32x8 acc[4] = {};

  const __bf16* aRow = A + (long)(row0 + rlo) * K + kHalf;
  for (int k0 = 0; k0 < K; k0 += 32) {
    union { bf16x16 v; bf16x8 h[2]; } a;
    a.h[0] = *(const bf16x8*)(aRow + k0);        // K = k0+kHalf .. +7
    a.h[1] = *(const bf16x8*)(aRow + k0 + 16);   // K = k0+kHalf+16 .. +7
#pragma unroll
    for (int j = 0; j < 4; ++j) {
      union { bf16x16 v; bf16x8 h[2]; } b;
      const __bf16* bRow = Bt + (long)(col0 + j * 16 + rlo) * K + kHalf;
      b.h[0] = *(const bf16x8*)(bRow + k0);
      b.h[1] = *(const bf16x8*)(bRow + k0 + 16);
      acc[j] = __builtin_amdgcn_wmma_f32_16x16x32_bf16(
          /*neg_a=*/false, a.v, /*neg_b=*/false, b.v,
          /*c_mod=*/(short)0, acc[j], /*reuse_a=*/false, /*reuse_b=*/false);
    }
  }
  // C layout: VGPR i -> M = i + (lane>=16 ? 8 : 0), N = lane&15
  const int mBase = row0 + ((lane >> 4) << 3);
#pragma unroll
  for (int j = 0; j < 4; ++j) {
    const int n = col0 + j * 16 + rlo;
#pragma unroll
    for (int i = 0; i < 8; ++i)
      C[(long)(mBase + i) * C1 + n] = acc[j][i];
  }
}

// ---- per-node attention scores: es/ed [N,HEADS] ----
__global__ void k_attn_scores(const float* __restrict__ h,
                              const float* __restrict__ a_s,
                              const float* __restrict__ a_d,
                              float* __restrict__ es, float* __restrict__ ed) {
  int i = blockIdx.x * blockDim.x + threadIdx.x;
  if (i >= NN * HEADS) return;
  int n = i >> 1, hd = i & 1;
  const float* hp = h + (long)n * C1 + hd * HID;
  const float* as = a_s + hd * HID;
  const float* ad = a_d + hd * HID;
  float s = 0.f, d = 0.f;
  for (int c = 0; c < HID; ++c) { float v = hp[c]; s += v * as[c]; d += v * ad[c]; }
  es[i] = s; ed[i] = d;
}

// ---- init segment buffers + zero aggregation buffer ----
__global__ void k_init_seg(float* __restrict__ mseg, float* __restrict__ dseg,
                           float* __restrict__ agg) {
  long i = (long)blockIdx.x * blockDim.x + threadIdx.x;
  if (i < NN * HEADS) { mseg[i] = -__builtin_inff(); dseg[i] = 0.f; }
  long total = (long)NPAD * C1;
  long stride = (long)gridDim.x * blockDim.x;
  for (long j = i; j < total; j += stride) agg[j] = 0.f;
}

// ---- edge pass A: e = leaky_relu(es[src]+ed[dst]); segment max ----
__global__ void k_edge_max(const int* __restrict__ ei,
                           const float* __restrict__ es, const float* __restrict__ ed,
                           float* __restrict__ eedge, float* __restrict__ mseg) {
  int i = blockIdx.x * blockDim.x + threadIdx.x;
  if (i >= ETOT) return;
  int e = i >> 1, hd = i & 1;
  int s, d;
  if (e < EE) { s = ei[e]; d = ei[EE + e]; } else { s = d = e - EE; }
  float v = es[s * HEADS + hd] + ed[d * HEADS + hd];
  v = (v > 0.f) ? v : NEG_SLOPE * v;
  eedge[i] = v;
  atomicMaxF(&mseg[d * HEADS + hd], v);
}

// ---- edge pass B: ex = exp(e - m[dst]); segment sum ----
__global__ void k_edge_exp(const int* __restrict__ ei,
                           const float* __restrict__ mseg,
                           float* __restrict__ eedge, float* __restrict__ dseg) {
  int i = blockIdx.x * blockDim.x + threadIdx.x;
  if (i >= ETOT) return;
  int e = i >> 1, hd = i & 1;
  int d = (e < EE) ? ei[EE + e] : (e - EE);
  float ex = __expf(eedge[i] - mseg[d * HEADS + hd]);
  eedge[i] = ex;
  atomicAdd(&dseg[d * HEADS + hd], ex);
}

// ---- edge pass C: agg[dst] += h[src] * alpha ; one wave per (edge,head) ----
__global__ void __launch_bounds__(256)
k_edge_aggr(const int* __restrict__ ei, const float* __restrict__ hfeat,
            const float* __restrict__ eedge, const float* __restrict__ dseg,
            float* __restrict__ agg) {
  int gw = (int)(((long)blockIdx.x * blockDim.x + threadIdx.x) >> 5);
  int lane = threadIdx.x & 31;
  if (gw >= ETOT) return;
  int e = gw >> 1, hd = gw & 1;
  int s, d;
  if (e < EE) { s = ei[e]; d = ei[EE + e]; } else { s = d = e - EE; }
  float alpha = eedge[gw] / dseg[d * HEADS + hd];
  const float* hs = hfeat + (long)s * C1 + hd * HID;
  float* ag = agg + (long)d * C1 + hd * HID;
#pragma unroll
  for (int j = 0; j < 4; ++j) {
    int c = lane + 32 * j;
    atomicAdd(&ag[c], hs[c] * alpha);
  }
}

// ---- layer-1 finalize: relu(agg + b1) -> bf16 (padded rows = 0) ----
__global__ void k_finalize_relu(const float* __restrict__ agg,
                                const float* __restrict__ bias,
                                __bf16* __restrict__ out_b) {
  long i = (long)blockIdx.x * blockDim.x + threadIdx.x;
  long tot = (long)NPAD * C1;
  if (i >= tot) return;
  long r = i / C1, c = i % C1;
  float v = 0.f;
  if (r < NN) { v = agg[i] + bias[c]; v = v > 0.f ? v : 0.f; }
  out_b[i] = f2bf(v);
}

// ---- layer-2 finalize + node mean pool: pooled[bt,c] = mean_n(agg)+b2[c] ----
__global__ void __launch_bounds__(256)
k_pool(const float* __restrict__ agg, const float* __restrict__ bias,
       float* __restrict__ pooled, int bt) {
  int c = blockIdx.x;                   // 0..255
  __shared__ float red[256];
  float s = 0.f;
  for (int n = threadIdx.x; n < NN; n += 256) s += agg[(long)n * C1 + c];
  red[threadIdx.x] = s;
  __syncthreads();
  for (int off = 128; off > 0; off >>= 1) {
    if (threadIdx.x < off) red[threadIdx.x] += red[threadIdx.x + off];
    __syncthreads();
  }
  if (threadIdx.x == 0) pooled[(long)bt * C1 + c] = red[0] / (float)NN + bias[c];
}

// ---- GRU over T steps, one block per batch; outputs time-mean hidden ----
__global__ void __launch_bounds__(384)
k_gru(const float* __restrict__ pooled,
      const float* __restrict__ Wih, const float* __restrict__ Whh,
      const float* __restrict__ bih, const float* __restrict__ bhh,
      float* __restrict__ gsum) {
  int b = blockIdx.x;
  int j = threadIdx.x;                  // 0..383
  __shared__ float gi[3 * HID], gh[3 * HID], hcur[HID], hsum[HID];
  if (j < HID) { hcur[j] = 0.f; hsum[j] = 0.f; }
  __syncthreads();
  for (int t = 0; t < TT; ++t) {
    const float* x = pooled + ((long)b * TT + t) * C1;
    float si = bih[j], sh = bhh[j];
    const float* wi = Wih + (long)j * C1;
    for (int k = 0; k < C1; ++k) si += x[k] * wi[k];
    const float* wh = Whh + (long)j * HID;
    for (int k = 0; k < HID; ++k) sh += hcur[k] * wh[k];
    gi[j] = si; gh[j] = sh;
    __syncthreads();
    if (j < HID) {
      float r  = 1.f / (1.f + __expf(-(gi[j] + gh[j])));
      float z  = 1.f / (1.f + __expf(-(gi[HID + j] + gh[HID + j])));
      float nn = tanhf(gi[2 * HID + j] + r * gh[2 * HID + j]);
      float hn = (1.f - z) * nn + z * hcur[j];
      hcur[j] = hn; hsum[j] += hn;
    }
    __syncthreads();
  }
  if (j < HID) gsum[b * HID + j] = hsum[j] / (float)TT;
}

// ---- classifier: out[b,o] = gsum[b]@Wc[:,o] + bc[o] ----
__global__ void k_classify(const float* __restrict__ gsum,
                           const float* __restrict__ Wc, const float* __restrict__ bc,
                           float* __restrict__ out) {
  int i = blockIdx.x * blockDim.x + threadIdx.x;
  if (i >= BQ * OUTC) return;
  int b = i / OUTC, o = i % OUTC;
  float s = bc[o];
  for (int k = 0; k < HID; ++k) s += gsum[b * HID + k] * Wc[k * OUTC + o];
  out[i] = s;
}

static inline int cdiv(long a, long b) { return (int)((a + b - 1) / b); }

extern "C" void kernel_launch(void* const* d_in, const int* in_sizes, int n_in,
                              void* d_out, int out_size, void* d_ws, size_t ws_size,
                              hipStream_t stream) {
  // inputs in setup_inputs() dict order
  const float* x_seq  = (const float*)d_in[0];   // [B,T,N,FIN]
  const int*   eidx   = (const int*)  d_in[1];   // [T,2,E]
  const float* W1     = (const float*)d_in[2];   // [FIN, 2*HID]
  const float* a_src1 = (const float*)d_in[3];
  const float* a_dst1 = (const float*)d_in[4];
  const float* b1     = (const float*)d_in[5];
  const float* W2     = (const float*)d_in[6];   // [2*HID, 2*HID]
  const float* a_src2 = (const float*)d_in[7];
  const float* a_dst2 = (const float*)d_in[8];
  const float* b2     = (const float*)d_in[9];
  const float* Wih    = (const float*)d_in[10];  // [3*HID, 2*HID]
  const float* Whh    = (const float*)d_in[11];  // [3*HID, HID]
  const float* bih    = (const float*)d_in[12];
  const float* bhh    = (const float*)d_in[13];
  const float* Wc     = (const float*)d_in[14];  // [HID, OUT]
  const float* bc     = (const float*)d_in[15];
  float* out = (float*)d_out;                    // [B, OUT]

  // ---- workspace carve-out ----
  char* ws = (char*)d_ws;
  size_t off = 0;
  auto carve = [&](size_t bytes) -> char* {
    char* p = ws + off;
    off = (off + bytes + 255) & ~((size_t)255);
    return p;
  };
  __bf16* Wt1   = (__bf16*)carve((size_t)C1 * FIN * 2);     // [256,128]
  __bf16* Wt2   = (__bf16*)carve((size_t)C1 * C1 * 2);      // [256,256]
  __bf16* xb    = (__bf16*)carve((size_t)NPAD * FIN * 2);
  __bf16* hb    = (__bf16*)carve((size_t)NPAD * C1 * 2);
  float*  hfeat = (float*) carve((size_t)NPAD * C1 * 4);
  float*  agg   = (float*) carve((size_t)NPAD * C1 * 4);
  float*  es    = (float*) carve((size_t)NN * HEADS * 4);
  float*  ed    = (float*) carve((size_t)NN * HEADS * 4);
  float*  mseg  = (float*) carve((size_t)NN * HEADS * 4);
  float*  dseg  = (float*) carve((size_t)NN * HEADS * 4);
  float*  eedge = (float*) carve((size_t)ETOT * 4);
  float*  pooled= (float*) carve((size_t)BQ * TT * C1 * 4);
  float*  gsum  = (float*) carve((size_t)BQ * HID * 4);
  (void)ws_size; (void)in_sizes; (void)n_in; (void)out_size;

  // ---- weight transpose+convert (once per call) ----
  k_transpose_w<<<cdiv((long)FIN * C1, 256), 256, 0, stream>>>(W1, Wt1, FIN, C1);
  k_transpose_w<<<cdiv((long)C1 * C1, 256), 256, 0, stream>>>(W2, Wt2, C1, C1);

  const dim3 gemmGrid(NPAD / 128, C1 / 64);
  const int convBlocks1 = cdiv((long)NPAD * FIN, 256);
  const int convBlocks2 = cdiv((long)NPAD * C1, 256);
  const int edgeBlocks  = cdiv((long)ETOT, 256);
  const int aggrBlocks  = cdiv((long)ETOT * 32, 256);

  for (int b = 0; b < BQ; ++b) {
    for (int t = 0; t < TT; ++t) {
      const float* xt  = x_seq + ((long)b * TT + t) * (long)NN * FIN;
      const int*   eit = eidx + (long)t * 2 * EE;

      // ===== GAT layer 1 =====
      k_convert_x<<<convBlocks1, 256, 0, stream>>>(xt, xb);
      k_gemm_bf16<<<gemmGrid, 256, 0, stream>>>(xb, Wt1, hfeat, FIN);
      k_attn_scores<<<cdiv((long)NN * HEADS, 256), 256, 0, stream>>>(hfeat, a_src1, a_dst1, es, ed);
      k_init_seg<<<128, 256, 0, stream>>>(mseg, dseg, agg);
      k_edge_max<<<edgeBlocks, 256, 0, stream>>>(eit, es, ed, eedge, mseg);
      k_edge_exp<<<edgeBlocks, 256, 0, stream>>>(eit, mseg, eedge, dseg);
      k_edge_aggr<<<aggrBlocks, 256, 0, stream>>>(eit, hfeat, eedge, dseg, agg);
      k_finalize_relu<<<convBlocks2, 256, 0, stream>>>(agg, b1, hb);

      // ===== GAT layer 2 =====
      k_gemm_bf16<<<gemmGrid, 256, 0, stream>>>(hb, Wt2, hfeat, C1);
      k_attn_scores<<<cdiv((long)NN * HEADS, 256), 256, 0, stream>>>(hfeat, a_src2, a_dst2, es, ed);
      k_init_seg<<<128, 256, 0, stream>>>(mseg, dseg, agg);
      k_edge_max<<<edgeBlocks, 256, 0, stream>>>(eit, es, ed, eedge, mseg);
      k_edge_exp<<<edgeBlocks, 256, 0, stream>>>(eit, mseg, eedge, dseg);
      k_edge_aggr<<<aggrBlocks, 256, 0, stream>>>(eit, hfeat, eedge, dseg, agg);
      k_pool<<<C1, 256, 0, stream>>>(agg, b2, pooled, b * TT + t);
    }
  }

  // ===== GRU + mean + classifier =====
  k_gru<<<BQ, 384, 0, stream>>>(pooled, Wih, Whh, bih, bhh, gsum);
  k_classify<<<1, 64, 0, stream>>>(gsum, Wc, bc, out);
}
